// VQVAE_7825430413749
// MI455X (gfx1250) — matrix operand order repaired and checked by reading, not compiled, not run on hardware
//
#include <hip/hip_runtime.h>

// ---------------------------------------------------------------------------
// VQ-VAE forward for MI455X (gfx1250): bf16 WMMA implicit-GEMM convolutions,
// fp32 accumulation, NHWC activations, double-buffered LDS pipeline fed by
// the Tensor Data Mover (weights, pre-permuted to WMMA fragment order) and
// per-lane async global->LDS copies (im2col, scattered to fragment order).
// All LDS operand fetches are contiguous per-lane ds_load_b128.
// ---------------------------------------------------------------------------

typedef __bf16 bf16;
typedef __attribute__((ext_vector_type(16))) __bf16 v16bf;
typedef __attribute__((ext_vector_type(16))) float  v16f;
typedef __attribute__((ext_vector_type(8)))  float  v8f;
typedef __attribute__((ext_vector_type(4)))  unsigned int u32x4;
typedef __attribute__((ext_vector_type(8)))  unsigned int u32x8;

#define TMB 64     // block tile M (2 waves x 32 rows)
#define TNB 128    // block tile N (4 waves x 32 cols)
#define KBLK 64    // K staged per iteration (2 WMMA K-substeps of 32)

// ---------------------------------------------------------------------------
// Implicit-GEMM conv / conv_t / GEMM kernel.
//   out[m, co] = alpha * sum_k A[m,k]*B[k,co] + bias[co]
//   A: im2col of NHWC fp32 input, scattered into LDS in WMMA-fragment order
//      by global_load_async_to_lds_b32 (OOB lanes read a zeroed dword so
//      EXEC stays all-ones).
//   B: weights packed in global memory as [k/32][col][khalf][16] bf16
//      (zero-padded to K%64==0), so tensor_load_to_lds' linear write lands
//      in fragment order; tensor dims are relative to the tile start so TDM
//      OOB-returns-zero implements the col<Cout guard.
//   ldil > 1 implements ConvTranspose2d as an lhs-dilated convolution.
// ---------------------------------------------------------------------------
__global__ __launch_bounds__(256) void conv_wmma_kernel(
    const float* __restrict__ in, const bf16* __restrict__ wpk,
    const float* __restrict__ bias, float* __restrict__ out,
    const float* __restrict__ zflt,
    int Nb, int Hin, int Win, int Cin, int Hout, int Wout, int Cout,
    int KH, int KW, int stride, int pad, int ldil, float alpha)
{
  const int M    = Nb * Hout * Wout;
  const int Ktot = KH * KW * Cin;
  const int mBase = blockIdx.x * TMB;
  const int nBase = blockIdx.y * TNB;

  // Fragment-major LDS:
  //  Asf: [buf][m16(4)][k32(2)][lane(32)][e(16)] fp32    (32 KB)
  //  Bs : [buf][k32(2)][col(128)][khalf(2)][e(16)] bf16  (32 KB)
  __shared__ alignas(64) float Asf[2][TMB * KBLK];
  __shared__ alignas(64) bf16  Bs [2][KBLK * TNB];

  const int tid    = threadIdx.x;
  const int lane   = tid & 31;
  const int wave   = tid >> 5;      // 0..7 (wave32)
  const int wm     = wave & 1;      // M strip (32 rows)
  const int wn     = wave >> 1;     // N strip (32 cols)
  const int laneHi = (lane >> 4) & 1;
  const int lane15 = lane & 15;

  // LDS byte offsets: flat aperture maps LDS as addr[31:0].
  const unsigned ldsA_base = (unsigned)(unsigned long long)(const void*)&Asf[0][0];
  const unsigned ldsB_base = (unsigned)(unsigned long long)(const void*)&Bs[0][0];

  __builtin_prefetch(bias + nBase, 0, 1);   // global_prefetch_b8

  v8f acc00 = {}, acc01 = {}, acc10 = {}, acc11 = {};

  const int ksteps = (Ktot + KBLK - 1) / KBLK;
  const int Kpad   = ksteps * KBLK;          // weight buffer is padded to this

  // ---- stage one K-block into LDS buffer `buf` (fire-and-forget DMA) ----
  auto stage = [&](int buf, int k0) {
    if (wave == 0) {   // TDM: 2 lines x (128 cols * 32 values) bf16, permuted
      const int kb0 = k0 >> 5;   // 32-wide K block index
      const unsigned long long ga = (unsigned long long)(const void*)wpk +
          ((unsigned long long)((unsigned)kb0 * (unsigned)Cout + (unsigned)nBase) * 32u) *
              sizeof(bf16);
      const unsigned td0 = (unsigned)(Cout - nBase) * 32u;  // remaining elems/line
      const unsigned td1 = (unsigned)((Kpad - k0) >> 5);    // remaining K lines
      u32x4 g0;
      g0[0] = 1u;                                      // count=1, user descriptor
      g0[1] = ldsB_base + (unsigned)buf * (KBLK * TNB * 2u);
      g0[2] = (unsigned)(ga & 0xFFFFFFFFull);          // global_addr[31:0]
      g0[3] = (unsigned)((ga >> 32) & 0x01FFFFFFull)   // global_addr[56:32]
            | (2u << 30);                              // type = image
      u32x8 g1;
      g1[0] = 1u << 16;                                // data_size = 2 bytes
      g1[1] = (td0 & 0xFFFFu) << 16;                   // tensor_dim0[15:0]
      g1[2] = ((td0 >> 16) & 0xFFFFu) | ((td1 & 0xFFFFu) << 16);
      g1[3] = ((td1 >> 16) & 0xFFFFu)
            | ((unsigned)(TNB * 32) << 16);            // tile_dim0 = 4096
      g1[4] = 2u;                                      // tile_dim1 = 2 lines
      g1[5] = (unsigned)Cout * 32u;                    // tensor_dim0_stride
      g1[6] = 0u;
      g1[7] = 0u;
      asm volatile("tensor_load_to_lds %0, %1" :: "s"(g0), "s"(g1) : "memory");
    }
    // Async per-lane im2col, scattered directly into WMMA fragment order:
    //   element (row r, k kk) -> lane ((kf>>4)&1)*16 + (r&15),
    //                            e = kf&15.
    for (int i = tid; i < TMB * KBLK; i += 256) {
      const int r  = i >> 6;          // KBLK == 64
      const int kk = i & 63;
      const int k  = k0 + kk;
      const int m  = mBase + r;
      const float* src = zflt;        // zero dword for padding / OOB
      if (m < M && k < Ktot) {
        const int ci = k % Cin;
        int t        = k / Cin;
        const int kx = t % KW;
        const int ky = t / KW;
        const int ox = m % Wout;
        int t2       = m / Wout;
        const int oy = t2 % Hout;
        const int nb = t2 / Hout;
        const int yp = oy * stride - pad + ky;
        const int xp = ox * stride - pad + kx;
        int iy = yp, ix = xp;
        bool ok = (yp >= 0) && (xp >= 0);
        if (ldil > 1) {
          ok = ok && ((yp % ldil) == 0) && ((xp % ldil) == 0);
          iy = yp / ldil; ix = xp / ldil;
        }
        ok = ok && (iy < Hin) && (ix < Win);
        if (ok) src = &in[(((size_t)nb * Hin + iy) * Win + ix) * Cin + ci];
      }
      const int m16 = r >> 4, mr = r & 15;
      const int k32 = kk >> 5, kf = kk & 31;
      const int lp  = ((kf >> 4) & 1) * 16 + mr;
      const int e   = kf & 15;
      const unsigned idx = (((unsigned)(m16 * 2 + k32) * 32 + (unsigned)lp) * 16
                            + (unsigned)e);
      const unsigned lds = ldsA_base + ((unsigned)buf * (TMB * KBLK) + idx) * 4u;
      const unsigned long long ga = (unsigned long long)src;
      asm volatile("global_load_async_to_lds_b32 %0, %1, off"
                   :: "v"(lds), "v"(ga) : "memory");
    }
  };

  stage(0, 0);
  asm volatile("s_wait_asynccnt 0x0" ::: "memory");
  if (wave == 0) __builtin_amdgcn_s_wait_tensorcnt(0);
  __syncthreads();

  const int bcol0 = wn * 32 + lane15;
  const int bcol1 = bcol0 + 16;

  for (int ks = 0; ks < ksteps; ++ks) {
    const int cur = ks & 1;
    if (ks + 1 < ksteps) stage(cur ^ 1, (ks + 1) * KBLK);  // overlap with math

    // ---- 2 K-substeps x (2 M-frags x 2 N-frags) = 8 WMMAs ----
    #pragma unroll
    for (int k32 = 0; k32 < 2; ++k32) {
      // A fragments: 64B contiguous per lane -> 4x ds_load_b128 each + cvt.
      const v16f va0 = *(const v16f*)&Asf[cur][(((wm * 2 + 0) * 2 + k32) * 32 + lane) * 16];
      const v16f va1 = *(const v16f*)&Asf[cur][(((wm * 2 + 1) * 2 + k32) * 32 + lane) * 16];
      v16bf a0, a1;
      #pragma unroll
      for (int e = 0; e < 16; ++e) { a0[e] = (bf16)va0[e]; a1[e] = (bf16)va1[e]; }
      // B fragments: 32B contiguous per lane -> 2x ds_load_b128 each.
      const v16bf b0 = *(const v16bf*)&Bs[cur][((k32 * TNB + bcol0) * 2 + laneHi) * 16];
      const v16bf b1 = *(const v16bf*)&Bs[cur][((k32 * TNB + bcol1) * 2 + laneHi) * 16];

      acc00 = __builtin_amdgcn_wmma_f32_16x16x32_bf16(false, a0, false, b0,
                                                      (short)0, acc00, false, false);
      acc01 = __builtin_amdgcn_wmma_f32_16x16x32_bf16(false, a0, false, b1,
                                                      (short)0, acc01, false, false);
      acc10 = __builtin_amdgcn_wmma_f32_16x16x32_bf16(false, a1, false, b0,
                                                      (short)0, acc10, false, false);
      acc11 = __builtin_amdgcn_wmma_f32_16x16x32_bf16(false, a1, false, b1,
                                                      (short)0, acc11, false, false);
    }

    asm volatile("s_wait_asynccnt 0x0" ::: "memory");    // our staged A loads
    if (wave == 0) __builtin_amdgcn_s_wait_tensorcnt(0); // our staged B tile
    __syncthreads();
  }

  // ---- epilogue: C/D layout VGPR r -> row r (lanes 0-15) / r+8 (16-31) ----
  const int col0 = nBase + wn * 32 + lane15;
  const int col1 = col0 + 16;
  #pragma unroll
  for (int r = 0; r < 8; ++r) {
    const int mrow0 = mBase + wm * 32 + laneHi * 8 + r;
    const int mrow1 = mrow0 + 16;
    if (mrow0 < M) {
      if (col0 < Cout) out[(size_t)mrow0 * Cout + col0] = alpha * acc00[r] + bias[col0];
      if (col1 < Cout) out[(size_t)mrow0 * Cout + col1] = alpha * acc01[r] + bias[col1];
    }
    if (mrow1 < M) {
      if (col0 < Cout) out[(size_t)mrow1 * Cout + col0] = alpha * acc10[r] + bias[col0];
      if (col1 < Cout) out[(size_t)mrow1 * Cout + col1] = alpha * acc11[r] + bias[col1];
    }
  }
}

// ---------------------------------------------------------------------------
// Weight packing into WMMA-fragment-major order:
//   out[((kb*Cout + co)*2 + khalf)*16 + kr],  k = kb*32 + khalf*16 + kr,
// zero-padded to Kpad (multiple of 64). Normal convs read OIHW; tflip reads
// transposed-conv IOHW with spatial flip. K = (ky*KW + kx)*Cin + ci.
// ---------------------------------------------------------------------------
__global__ void pack_w_kernel(const float* __restrict__ w, bf16* __restrict__ out,
                              int Cout, int Cin, int KH, int KW, int tflip,
                              int Ktot, int Kpad)
{
  const int i = blockIdx.x * 256 + threadIdx.x;
  const int total = (Kpad >> 5) * Cout * 32;
  if (i >= total) return;
  const int kr   = i & 15;
  int t          = i >> 4;
  const int half = t & 1;  t >>= 1;
  const int co   = t % Cout;
  const int kb   = t / Cout;
  const int k    = kb * 32 + half * 16 + kr;
  bf16 v = (bf16)0.0f;
  if (k < Ktot) {
    const int ci = k % Cin;
    int q        = k / Cin;
    const int kx = q % KW;
    const int ky = q / KW;
    float f;
    if (!tflip) f = w[(((size_t)co * Cin + ci) * KH + ky) * KW + kx];
    else        f = w[(((size_t)ci * Cout + co) * KH + (KH - 1 - ky)) * KW + (KW - 1 - kx)];
    v = (bf16)f;
  }
  out[i] = v;
}

__global__ void zero_init_kernel(float* __restrict__ p, int n)
{
  const int i = blockIdx.x * 64 + threadIdx.x;
  if (i < n) p[i] = 0.0f;
}

__global__ void dnorm_kernel(const float* __restrict__ dic, float* __restrict__ dn,
                             int F, int K)
{
  const int k = blockIdx.x * 256 + threadIdx.x;
  if (k >= K) return;
  float s = 0.0f;
  for (int f = 0; f < F; ++f) { const float v = dic[(size_t)f * K + k]; s += v * v; }
  dn[k] = s;
}

// ---------------------------------------------------------------------------
// BatchNorm (training-mode batch statistics) on NHWC: one block per channel.
// ---------------------------------------------------------------------------
__global__ __launch_bounds__(256) void bn_stats_kernel(
    const float* __restrict__ x, float* __restrict__ mean, float* __restrict__ var,
    int M, int C)
{
  const int c = blockIdx.x;
  __shared__ float ss[256];
  __shared__ float ss2[256];
  float s = 0.0f, s2 = 0.0f;
  for (int i = threadIdx.x; i < M; i += 256) {
    const float v = x[(size_t)i * C + c];
    s += v; s2 += v * v;
  }
  ss[threadIdx.x] = s; ss2[threadIdx.x] = s2;
  __syncthreads();
  for (int o = 128; o > 0; o >>= 1) {
    if (threadIdx.x < o) {
      ss[threadIdx.x]  += ss[threadIdx.x + o];
      ss2[threadIdx.x] += ss2[threadIdx.x + o];
    }
    __syncthreads();
  }
  if (threadIdx.x == 0) {
    const float m = ss[0] / (float)M;
    mean[c] = m;
    var[c]  = ss2[0] / (float)M - m * m;   // biased, matches jnp.var
  }
}

__global__ void bn_apply_kernel(const float* __restrict__ x,
                                const float* __restrict__ mean,
                                const float* __restrict__ var,
                                const float* __restrict__ g,
                                const float* __restrict__ be,
                                float* __restrict__ out,
                                int M, int C, int relu)
{
  const size_t i = (size_t)blockIdx.x * 256 + threadIdx.x;
  if (i >= (size_t)M * C) return;
  const int c = (int)(i % C);
  float v = (x[i] - mean[c]) * rsqrtf(var[c] + 1e-5f) * g[c] + be[c];
  if (relu) v = fmaxf(v, 0.0f);
  out[i] = v;
}

__global__ void relu_kernel(const float* __restrict__ in, float* __restrict__ out,
                            size_t n)
{
  const size_t i = (size_t)blockIdx.x * 256 + threadIdx.x;
  if (i < n) out[i] = fmaxf(in[i], 0.0f);
}

__global__ void add_kernel(const float* __restrict__ a, const float* __restrict__ b,
                           float* __restrict__ out, size_t n)
{
  const size_t i = (size_t)blockIdx.x * 256 + threadIdx.x;
  if (i < n) out[i] = a[i] + b[i];
}

__global__ void nchw2nhwc_kernel(const float* __restrict__ in, float* __restrict__ out,
                                 int N, int C, int H, int W)
{
  const size_t i = (size_t)blockIdx.x * 256 + threadIdx.x;
  const size_t total = (size_t)N * C * H * W;
  if (i >= total) return;
  const int x = (int)(i % W); size_t t = i / W;
  const int y = (int)(t % H); t /= H;
  const int c = (int)(t % C); const int n = (int)(t / C);
  out[(((size_t)n * H + y) * W + x) * C + c] = in[i];
}

__global__ void nhwc2nchw_kernel(const float* __restrict__ in, float* __restrict__ out,
                                 int N, int H, int W, int C, int do_tanh)
{
  const size_t i = (size_t)blockIdx.x * 256 + threadIdx.x;
  const size_t total = (size_t)N * H * W * C;
  if (i >= total) return;
  const int c = (int)(i % C); size_t t = i / C;
  const int x = (int)(t % W); t /= W;
  const int y = (int)(t % H); const int n = (int)(t / H);
  float v = in[i];
  if (do_tanh) v = tanhf(v);
  out[(((size_t)n * C + c) * H + y) * W + x] = v;
}

// Per-pixel argmin over K codebook scores + gather of selected code.
__global__ void vq_argmin_kernel(const float* __restrict__ scores,
                                 const float* __restrict__ dic,
                                 float* __restrict__ vqz_nhwc,
                                 float* __restrict__ idx_out,
                                 int M, int K, int F)
{
  const int i = blockIdx.x * 256 + threadIdx.x;
  if (i >= M) return;
  float best = scores[(size_t)i * K];
  int bi = 0;
  for (int k = 1; k < K; ++k) {
    const float s = scores[(size_t)i * K + k];
    if (s < best) { best = s; bi = k; }
  }
  idx_out[i] = (float)bi;
  for (int f = 0; f < F; ++f)
    vqz_nhwc[(size_t)i * F + f] = dic[(size_t)f * K + bi];
}

// ---------------------------------------------------------------------------
// Host-side orchestration
// ---------------------------------------------------------------------------
static inline int cdiv(int a, int b) { return (a + b - 1) / b; }
static inline int kpad64(int k) { return cdiv(k, 64) * 64; }
static inline unsigned gridFor(size_t n) { return (unsigned)((n + 255) / 256); }

static const float* g_zflt = nullptr;   // set every kernel_launch (no state kept)

static void launch_conv(hipStream_t s, const float* in, const bf16* w, const float* bias,
                        float* out, int Nb, int Hin, int Win, int Cin,
                        int Hout, int Wout, int Cout,
                        int KH, int KW, int stride, int pad, int ldil, float alpha)
{
  const int M = Nb * Hout * Wout;
  dim3 g(cdiv(M, TMB), cdiv(Cout, TNB));
  conv_wmma_kernel<<<g, 256, 0, s>>>(in, w, bias, out, g_zflt, Nb, Hin, Win, Cin,
                                     Hout, Wout, Cout, KH, KW, stride, pad, ldil, alpha);
}

static void launch_bn(hipStream_t s, const float* x, float* out, const float* g,
                      const float* be, float* mean, float* var, int M, int C, int relu)
{
  bn_stats_kernel<<<C, 256, 0, s>>>(x, mean, var, M, C);
  bn_apply_kernel<<<gridFor((size_t)M * C), 256, 0, s>>>(x, mean, var, g, be, out, M, C, relu);
}

struct RBp {
  const bf16* pw1; const float* b1; const float* g1; const float* be1;
  const bf16* pw2; const float* b2; const float* g2; const float* be2;
};

// X holds the block input (updated in place); T, U are scratch of same size.
static void launch_resblock(hipStream_t s, float* X, float* T, float* U, const RBp& p,
                            float* mean, float* var, int Nb, int H, int W)
{
  const int C = 128;
  const int M = Nb * H * W;
  const size_t total = (size_t)M * C;
  relu_kernel<<<gridFor(total), 256, 0, s>>>(X, T, total);
  launch_conv(s, T, p.pw1, p.b1, U, Nb, H, W, C, H, W, C, 3, 3, 1, 1, 1, 1.0f);
  launch_bn(s, U, T, p.g1, p.be1, mean, var, M, C, 1);
  launch_conv(s, T, p.pw2, p.b2, U, Nb, H, W, C, H, W, C, 1, 1, 1, 0, 1, 1.0f);
  launch_bn(s, U, T, p.g2, p.be2, mean, var, M, C, 0);
  add_kernel<<<gridFor(total), 256, 0, s>>>(X, T, X, total);
}

static void launch_pack(hipStream_t s, const float* w, bf16* out,
                        int Cout, int Cin, int KH, int KW, int tflip)
{
  const int Ktot = KH * KW * Cin;
  const int Kpad = kpad64(Ktot);
  const int total = Kpad * Cout;
  pack_w_kernel<<<cdiv(total, 256), 256, 0, s>>>(w, out, Cout, Cin, KH, KW, tflip,
                                                 Ktot, Kpad);
}

extern "C" void kernel_launch(void* const* d_in, const int* in_sizes, int n_in,
                              void* d_out, int out_size, void* d_ws, size_t ws_size,
                              hipStream_t stream)
{
  (void)in_sizes; (void)n_in; (void)out_size; (void)ws_size;
  const float** F = (const float**)d_in;

  // --- input pointer map (setup_inputs dict order, flattened) ---
  int ii = 0;
  const float* x     = F[ii++];                                 // 0
  const float* e_w0  = F[ii++]; const float* e_b0 = F[ii++];
  const float* e_g0  = F[ii++]; const float* e_be0 = F[ii++];   // 1..4
  const float* e_w1  = F[ii++]; const float* e_b1 = F[ii++];    // 5..6
  const float* er[2][8]; const float* dr[2][8];
  for (int r = 0; r < 2; ++r) for (int j = 0; j < 8; ++j) er[r][j] = F[ii++];  // 7..22
  const float* e_w2  = F[ii++]; const float* e_b2 = F[ii++];    // 23..24
  const float* d_w0  = F[ii++]; const float* d_b0 = F[ii++];
  const float* d_g0  = F[ii++]; const float* d_be0 = F[ii++];   // 25..28
  const float* d_w1  = F[ii++]; const float* d_b1 = F[ii++];    // 29..30
  for (int r = 0; r < 2; ++r) for (int j = 0; j < 8; ++j) dr[r][j] = F[ii++];  // 31..46
  const float* d_w2  = F[ii++]; const float* d_b2 = F[ii++];    // 47..48
  const float* dic   = F[ii++];                                 // 49

  // --- output regions (return order: xr, z, vqz, idx) ---
  float* out = (float*)d_out;
  const size_t n_xr = (size_t)8 * 3 * 224 * 224;
  const size_t n_z  = (size_t)8 * 64 * 56 * 56;
  float* o_xr  = out;
  float* o_z   = out + n_xr;
  float* o_vqz = out + n_xr + n_z;
  float* o_idx = out + n_xr + 2 * n_z;

  // --- workspace bump allocator ---
  size_t off = 0;
  char* base = (char*)d_ws;
  auto alloc = [&](size_t bytes) -> void* {
    void* p = base + off; off += (bytes + 255) & ~(size_t)255; return p;
  };
  const size_t S_BIG = (size_t)8 * 224 * 224 * 128 * sizeof(float);  // 205.5 MB
  float* A  = (float*)alloc(S_BIG);
  float* Bb = (float*)alloc(S_BIG);
  float* Cc = (float*)alloc(S_BIG);
  float* mean = (float*)alloc(128 * sizeof(float));
  float* var  = (float*)alloc(128 * sizeof(float));
  float* dnrm = (float*)alloc(512 * sizeof(float));
  float* zflt = (float*)alloc(64 * sizeof(float));
  auto wbuf = [&](int k, int n) { return (bf16*)alloc((size_t)kpad64(k) * n * sizeof(bf16)); };
  bf16* pw_e0 = wbuf(48, 128);
  bf16* pw_e1 = wbuf(2048, 128);
  bf16* pw_er[2][2] = { { wbuf(1152, 128), wbuf(128, 128) },
                        { wbuf(1152, 128), wbuf(128, 128) } };
  bf16* pw_e2 = wbuf(128, 64);
  bf16* pw_d0 = wbuf(1024, 128);
  bf16* pw_d1 = wbuf(2048, 128);
  bf16* pw_dr[2][2] = { { wbuf(1152, 128), wbuf(128, 128) },
                        { wbuf(1152, 128), wbuf(128, 128) } };
  bf16* pw_d2 = wbuf(128, 3);
  bf16* dicb  = wbuf(64, 512);

  hipStream_t st = stream;
  g_zflt = zflt;
  zero_init_kernel<<<1, 64, 0, st>>>(zflt, 64);

  // --- pack all weights to fragment-major bf16 [k/32][co][khalf][16] ---
  launch_pack(st, e_w0, pw_e0, 128, 3, 4, 4, 0);
  launch_pack(st, e_w1, pw_e1, 128, 128, 4, 4, 0);
  for (int r = 0; r < 2; ++r) {
    launch_pack(st, er[r][0], pw_er[r][0], 128, 128, 3, 3, 0);
    launch_pack(st, er[r][4], pw_er[r][1], 128, 128, 1, 1, 0);
    launch_pack(st, dr[r][0], pw_dr[r][0], 128, 128, 3, 3, 0);
    launch_pack(st, dr[r][4], pw_dr[r][1], 128, 128, 1, 1, 0);
  }
  launch_pack(st, e_w2, pw_e2, 64, 128, 1, 1, 0);
  launch_pack(st, d_w0, pw_d0, 128, 64, 4, 4, 1);   // ConvT: swap in/out + flip
  launch_pack(st, d_w1, pw_d1, 128, 128, 4, 4, 1);
  launch_pack(st, d_w2, pw_d2, 3, 128, 1, 1, 1);
  // Codebook (F=64, K=512) as a 1x1 "transposed conv" weight: B[f][k] = dic.
  launch_pack(st, dic, dicb, 512, 64, 1, 1, 1);
  dnorm_kernel<<<cdiv(512, 256), 256, 0, st>>>(dic, dnrm, 64, 512);

  // --- encoder ---
  nchw2nhwc_kernel<<<gridFor((size_t)8 * 3 * 224 * 224), 256, 0, st>>>(x, A, 8, 3, 224, 224);
  launch_conv(st, A, pw_e0, e_b0, Bb, 8, 224, 224, 3, 112, 112, 128, 4, 4, 2, 1, 1, 1.0f);
  launch_bn(st, Bb, A, e_g0, e_be0, mean, var, 8 * 112 * 112, 128, 1);
  launch_conv(st, A, pw_e1, e_b1, Bb, 8, 112, 112, 128, 56, 56, 128, 4, 4, 2, 1, 1, 1.0f);
  for (int r = 0; r < 2; ++r) {
    RBp p{ pw_er[r][0], er[r][1], er[r][2], er[r][3],
           pw_er[r][1], er[r][5], er[r][6], er[r][7] };
    launch_resblock(st, Bb, A, Cc, p, mean, var, 8, 56, 56);
  }
  launch_conv(st, Bb, pw_e2, e_b2, A, 8, 56, 56, 128, 56, 56, 64, 1, 1, 1, 0, 1, 1.0f);
  nhwc2nchw_kernel<<<gridFor(n_z), 256, 0, st>>>(A, o_z, 8, 56, 56, 64, 0);

  // --- VQ: scores = ||d||^2 - 2 z.d  (same WMMA GEMM, alpha=-2, bias=||d||^2)
  launch_conv(st, A, dicb, dnrm, Cc, 8, 56, 56, 64, 56, 56, 512, 1, 1, 1, 0, 1, -2.0f);
  vq_argmin_kernel<<<cdiv(8 * 56 * 56, 256), 256, 0, st>>>(Cc, dic, Bb, o_idx,
                                                           8 * 56 * 56, 512, 64);
  nhwc2nchw_kernel<<<gridFor(n_z), 256, 0, st>>>(Bb, o_vqz, 8, 56, 56, 64, 0);

  // --- decoder (ConvT = lhs-dilated conv: stride 1, pad k-1-p = 2, dil 2) ---
  launch_conv(st, Bb, pw_d0, d_b0, A, 8, 56, 56, 64, 112, 112, 128, 4, 4, 1, 2, 2, 1.0f);
  launch_bn(st, A, Cc, d_g0, d_be0, mean, var, 8 * 112 * 112, 128, 1);
  launch_conv(st, Cc, pw_d1, d_b1, A, 8, 112, 112, 128, 224, 224, 128, 4, 4, 1, 2, 2, 1.0f);
  for (int r = 0; r < 2; ++r) {
    RBp p{ pw_dr[r][0], dr[r][1], dr[r][2], dr[r][3],
           pw_dr[r][1], dr[r][5], dr[r][6], dr[r][7] };
    launch_resblock(st, A, Bb, Cc, p, mean, var, 8, 224, 224);
  }
  launch_conv(st, A, pw_d2, d_b2, Bb, 8, 224, 224, 128, 224, 224, 3, 1, 1, 1, 0, 1, 1.0f);
  nhwc2nchw_kernel<<<gridFor(n_xr), 256, 0, st>>>(Bb, o_xr, 8, 224, 224, 3, 1);
}
